// Transformer_14645838480104
// MI455X (gfx1250) — compile-verified, matrix operand order checked by reading, not run on hardware
//
#include <hip/hip_runtime.h>
#include <hip/hip_bf16.h>

typedef __attribute__((ext_vector_type(16))) __bf16 v16bf;
typedef __attribute__((ext_vector_type(8)))  __bf16 v8bf;
typedef __attribute__((ext_vector_type(8)))  float  v8f;

#define BB   4
#define SS   512
#define DD   512
#define HH   8
#define HD   64
#define MM   (BB*SS)     /* 2048 rows (batch*seq) */
#define DIO  128
#define NEGINF (-1e9f)
#define LPAD 40          /* LDS row stride: 80B, 16B-aligned, conflict-free for b128 */

// ---------------------------------------------------------------------------
// Complex GEMM via Karatsuba (3 WMMAs per tile instead of 4):
//   RR = Ar@Wr, II = Ai@Wi, SS_ = (Ar+Ai)@(Wr+Wi)
//   Cr = alpha*(RR - II) + br ;  Ci = alpha*(SS_ - RR - II) + bi ; opt. cReLU
// A:[MM,K] f32 planes; W^T packed bf16 planes [N,K]: wr, wi, wr+wi.
// Block = 4 waves, tile 64x64; A tile staged f32->bf16 in LDS (3 planes).
// ---------------------------------------------------------------------------
__global__ __launch_bounds__(128)
void cgemm_kernel(const float* __restrict__ Ar, const float* __restrict__ Ai,
                  const __bf16* __restrict__ WrT, const __bf16* __restrict__ WiT,
                  const __bf16* __restrict__ WsT,
                  const float* __restrict__ br, const float* __restrict__ bi,
                  float* __restrict__ Cr, float* __restrict__ Ci,
                  int N, int K, float alpha, int relu)
{
    __shared__ __align__(16) __bf16 lAr[64][LPAD];
    __shared__ __align__(16) __bf16 lAi[64][LPAD];
    __shared__ __align__(16) __bf16 lAs[64][LPAD];

    const int tid  = threadIdx.x;
    const int wv   = tid >> 5;
    const int lane = tid & 31;
    const int ln16 = lane & 15;
    const int hk   = lane >> 4;          // 0: K in {0..7,16..23}; 1: {8..15,24..31}
    const int Mb   = blockIdx.y * 64;
    const int n0   = blockIdx.x * 64 + wv * 16;

    v8f aRR[4], aII[4], aSS[4];
    for (int t = 0; t < 4; ++t)
        for (int r = 0; r < 8; ++r) { aRR[t][r] = 0.f; aII[t][r] = 0.f; aSS[t][r] = 0.f; }

    const int srow = tid >> 1;              // staging: 64 rows x 32 cols, 16 f32/thread
    const int scol = (tid & 1) * 16;

    for (int k0 = 0; k0 < K; k0 += 32) {
        {   // stage A tile (f32 -> bf16, plus sum plane) into LDS, b128 stores
            const float* pr = Ar + (size_t)(Mb + srow) * K + k0 + scol;
            const float* pi = Ai + (size_t)(Mb + srow) * K + k0 + scol;
            if (k0 + 32 < K) __builtin_prefetch(pr + 32, 0, 1);
            v16bf cr_, ci_, cs_;
            #pragma unroll
            for (int i = 0; i < 16; ++i) {
                const float fr = pr[i], fi = pi[i];
                cr_[i] = (__bf16)fr; ci_[i] = (__bf16)fi; cs_[i] = (__bf16)(fr + fi);
            }
            *(v8bf*)&lAr[srow][scol]     = ((v8bf*)&cr_)[0];
            *(v8bf*)&lAr[srow][scol + 8] = ((v8bf*)&cr_)[1];
            *(v8bf*)&lAi[srow][scol]     = ((v8bf*)&ci_)[0];
            *(v8bf*)&lAi[srow][scol + 8] = ((v8bf*)&ci_)[1];
            *(v8bf*)&lAs[srow][scol]     = ((v8bf*)&cs_)[0];
            *(v8bf*)&lAs[srow][scol + 8] = ((v8bf*)&cs_)[1];
        }
        __syncthreads();

        // B fragments from pre-packed bf16 W^T planes (16B-aligned vector loads)
        v16bf fBr, fBi, fBs;
        {
            const size_t ro = (size_t)(n0 + ln16) * K + k0 + hk * 8;
            const v8bf* pr = (const v8bf*)(WrT + ro);
            const v8bf* pi = (const v8bf*)(WiT + ro);
            const v8bf* ps = (const v8bf*)(WsT + ro);
            ((v8bf*)&fBr)[0] = pr[0]; ((v8bf*)&fBr)[1] = pr[2];   // +16 elems = +2 v8bf
            ((v8bf*)&fBi)[0] = pi[0]; ((v8bf*)&fBi)[1] = pi[2];
            ((v8bf*)&fBs)[0] = ps[0]; ((v8bf*)&fBs)[1] = ps[2];
        }
        #pragma unroll
        for (int mt = 0; mt < 4; ++mt) {
            const int rA = mt * 16 + ln16;
            const int kb = hk * 8;
            v16bf fAr, fAi, fAs;
            ((v8bf*)&fAr)[0] = *(const v8bf*)&lAr[rA][kb];
            ((v8bf*)&fAr)[1] = *(const v8bf*)&lAr[rA][kb + 16];
            ((v8bf*)&fAi)[0] = *(const v8bf*)&lAi[rA][kb];
            ((v8bf*)&fAi)[1] = *(const v8bf*)&lAi[rA][kb + 16];
            ((v8bf*)&fAs)[0] = *(const v8bf*)&lAs[rA][kb];
            ((v8bf*)&fAs)[1] = *(const v8bf*)&lAs[rA][kb + 16];
            aRR[mt] = __builtin_amdgcn_wmma_f32_16x16x32_bf16(false, fAr, false, fBr, (short)0, aRR[mt], false, false);
            aII[mt] = __builtin_amdgcn_wmma_f32_16x16x32_bf16(false, fAi, false, fBi, (short)0, aII[mt], false, false);
            aSS[mt] = __builtin_amdgcn_wmma_f32_16x16x32_bf16(false, fAs, false, fBs, (short)0, aSS[mt], false, false);
        }
        __syncthreads();
    }

    const int nn = n0 + ln16;
    #pragma unroll
    for (int mt = 0; mt < 4; ++mt) {
        #pragma unroll
        for (int r = 0; r < 8; ++r) {
            const int m = Mb + mt * 16 + 8 * hk + r;
            const float rr = aRR[mt][r], ii = aII[mt][r];
            float cr = alpha * (rr - ii) + br[nn];
            float ci = alpha * (aSS[mt][r] - rr - ii) + bi[nn];
            if (relu) { cr = fmaxf(cr, 0.f); ci = fmaxf(ci, 0.f); }
            Cr[(size_t)m * N + nn] = cr;
            Ci[(size_t)m * N + nn] = ci;
        }
    }
}

// ---------------------------------------------------------------------------
// Attention scores: S[bh,i,j] = alpha*(sum_d Qr*Kr - Qi*Ki)  (+causal mask)
// ---------------------------------------------------------------------------
__global__ __launch_bounds__(128)
void scores_kernel(const float* __restrict__ Qr, const float* __restrict__ Qi,
                   const float* __restrict__ Kr, const float* __restrict__ Ki,
                   float* __restrict__ Sc, int causal, float alpha)
{
    const int tid = threadIdx.x, wv = tid >> 5, lane = tid & 31;
    const int ln16 = lane & 15, hk = lane >> 4;
    const int bh = blockIdx.z, b = bh >> 3, h = bh & 7;
    const int i0 = blockIdx.y * 64;
    const int j0 = blockIdx.x * 64 + wv * 16;

    v8f a1[4], a2[4];
    for (int t = 0; t < 4; ++t) for (int r = 0; r < 8; ++r) { a1[t][r] = 0.f; a2[t][r] = 0.f; }

    const size_t rowK = (size_t)(b * SS + j0 + ln16) * DD + h * HD;
    for (int k0 = 0; k0 < HD; k0 += 32) {
        v16bf fKr, fKi;
        const float* pkr = Kr + rowK + k0 + hk * 8;
        const float* pki = Ki + rowK + k0 + hk * 8;
        #pragma unroll
        for (int i = 0; i < 8; ++i) {
            fKr[i] = (__bf16)pkr[i];  fKr[8 + i] = (__bf16)pkr[16 + i];
            fKi[i] = (__bf16)pki[i];  fKi[8 + i] = (__bf16)pki[16 + i];
        }
        #pragma unroll
        for (int mt = 0; mt < 4; ++mt) {
            const size_t rowQ = (size_t)(b * SS + i0 + mt * 16 + ln16) * DD + h * HD + k0 + hk * 8;
            v16bf fQr, fQi;
            #pragma unroll
            for (int i = 0; i < 8; ++i) {
                fQr[i] = (__bf16)Qr[rowQ + i];  fQr[8 + i] = (__bf16)Qr[rowQ + 16 + i];
                fQi[i] = (__bf16)Qi[rowQ + i];  fQi[8 + i] = (__bf16)Qi[rowQ + 16 + i];
            }
            a1[mt] = __builtin_amdgcn_wmma_f32_16x16x32_bf16(false, fQr, false, fKr, (short)0, a1[mt], false, false);
            a2[mt] = __builtin_amdgcn_wmma_f32_16x16x32_bf16(false, fQi, false, fKi, (short)0, a2[mt], false, false);
        }
    }
    const int j = j0 + ln16;
    #pragma unroll
    for (int mt = 0; mt < 4; ++mt) {
        #pragma unroll
        for (int r = 0; r < 8; ++r) {
            const int i = i0 + mt * 16 + 8 * hk + r;
            float s = alpha * (a1[mt][r] - a2[mt][r]);
            if (causal && (j > i)) s += NEGINF;
            Sc[((size_t)bh * SS + i) * SS + j] = s;
        }
    }
}

// ---------------------------------------------------------------------------
// Attention output: O[bh] = P (Sq x Sk) @ V[bh] (Sk x HD), both planes.
// ---------------------------------------------------------------------------
__global__ __launch_bounds__(128)
void attnout_kernel(const float* __restrict__ Sc,
                    const float* __restrict__ Vr, const float* __restrict__ Vi,
                    float* __restrict__ Tr, float* __restrict__ Ti)
{
    const int tid = threadIdx.x, wv = tid >> 5, lane = tid & 31;
    const int ln16 = lane & 15, hk = lane >> 4;
    const int bh = blockIdx.z, b = bh >> 3, h = bh & 7;
    const int i0 = blockIdx.y * 64;
    const int n0 = wv * 16;

    v8f aR[4], aI[4];
    for (int t = 0; t < 4; ++t) for (int r = 0; r < 8; ++r) { aR[t][r] = 0.f; aI[t][r] = 0.f; }

    for (int k0 = 0; k0 < SS; k0 += 32) {
        v16bf fVr, fVi;
        #pragma unroll
        for (int i = 0; i < 8; ++i) {
            const size_t r1 = (size_t)(b * SS + k0 + hk * 8 + i)      * DD + h * HD + n0 + ln16;
            const size_t r2 = (size_t)(b * SS + k0 + hk * 8 + 16 + i) * DD + h * HD + n0 + ln16;
            fVr[i] = (__bf16)Vr[r1];  fVr[8 + i] = (__bf16)Vr[r2];
            fVi[i] = (__bf16)Vi[r1];  fVi[8 + i] = (__bf16)Vi[r2];
        }
        #pragma unroll
        for (int mt = 0; mt < 4; ++mt) {
            const float* pp = Sc + ((size_t)bh * SS + i0 + mt * 16 + ln16) * SS + k0 + hk * 8;
            v16bf fP;
            #pragma unroll
            for (int i = 0; i < 8; ++i) { fP[i] = (__bf16)pp[i]; fP[8 + i] = (__bf16)pp[16 + i]; }
            aR[mt] = __builtin_amdgcn_wmma_f32_16x16x32_bf16(false, fP, false, fVr, (short)0, aR[mt], false, false);
            aI[mt] = __builtin_amdgcn_wmma_f32_16x16x32_bf16(false, fP, false, fVi, (short)0, aI[mt], false, false);
        }
    }
    #pragma unroll
    for (int mt = 0; mt < 4; ++mt) {
        #pragma unroll
        for (int r = 0; r < 8; ++r) {
            const int i = i0 + mt * 16 + 8 * hk + r;
            const size_t o = (size_t)(b * SS + i) * DD + h * HD + n0 + ln16;
            Tr[o] = aR[mt][r];
            Ti[o] = aI[mt][r];
        }
    }
}

// ---------------------------------------------------------------------------
// Row softmax over Sk=512; one wave32 per row.
// ---------------------------------------------------------------------------
__global__ __launch_bounds__(256)
void softmax_kernel(float* __restrict__ Sc)
{
    const int row  = blockIdx.x * 8 + (threadIdx.x >> 5);
    const int lane = threadIdx.x & 31;
    float* p = Sc + (size_t)row * SS;
    float v[16];
    float mx = -3.4e38f;
    #pragma unroll
    for (int i = 0; i < 16; ++i) { v[i] = p[lane + i * 32]; mx = fmaxf(mx, v[i]); }
    #pragma unroll
    for (int o = 16; o >= 1; o >>= 1) mx = fmaxf(mx, __shfl_xor(mx, o, 32));
    float sum = 0.f;
    #pragma unroll
    for (int i = 0; i < 16; ++i) { v[i] = expf(v[i] - mx); sum += v[i]; }
    #pragma unroll
    for (int o = 16; o >= 1; o >>= 1) sum += __shfl_xor(sum, o, 32);
    const float inv = 1.f / sum;
    #pragma unroll
    for (int i = 0; i < 16; ++i) p[lane + i * 32] = v[i] * inv;
}

// ---------------------------------------------------------------------------
// Complex layernorm with fused residual add; one wave per row of D=512.
// ---------------------------------------------------------------------------
__global__ __launch_bounds__(256)
void cln_kernel(const float* __restrict__ Ar, const float* __restrict__ Ai,
                const float* __restrict__ Rr, const float* __restrict__ Ri,
                const float* __restrict__ gr, const float* __restrict__ gi,
                const float* __restrict__ br, const float* __restrict__ bi,
                float* __restrict__ Or, float* __restrict__ Oi)
{
    const int row  = blockIdx.x * 8 + (threadIdx.x >> 5);
    const int lane = threadIdx.x & 31;
    const size_t base = (size_t)row * DD;
    float xr[16], xi[16];
    float sr = 0.f, si = 0.f;
    #pragma unroll
    for (int i = 0; i < 16; ++i) {
        const int c = lane + i * 32;
        xr[i] = Ar[base + c] + Rr[base + c];
        xi[i] = Ai[base + c] + Ri[base + c];
        sr += xr[i]; si += xi[i];
    }
    #pragma unroll
    for (int o = 16; o >= 1; o >>= 1) { sr += __shfl_xor(sr, o, 32); si += __shfl_xor(si, o, 32); }
    const float mr = sr * (1.f / DD), mi = si * (1.f / DD);
    float vs = 0.f;
    #pragma unroll
    for (int i = 0; i < 16; ++i) { xr[i] -= mr; xi[i] -= mi; vs += xr[i] * xr[i] + xi[i] * xi[i]; }
    #pragma unroll
    for (int o = 16; o >= 1; o >>= 1) vs += __shfl_xor(vs, o, 32);
    const float inv = rsqrtf(vs * (1.f / DD) + 1e-5f);
    #pragma unroll
    for (int i = 0; i < 16; ++i) {
        const int c = lane + i * 32;
        const float nr = xr[i] * inv, ni = xi[i] * inv;
        Or[base + c] = nr * gr[c] - ni * gi[c] + br[c];
        Oi[base + c] = nr * gi[c] + ni * gr[c] + bi[c];
    }
}

// ---------------------------------------------------------------------------
// Elementwise helpers
// ---------------------------------------------------------------------------
__global__ void pe_kernel(float* __restrict__ Xr, int posoff, int n)
{
    const int idx = blockIdx.x * blockDim.x + threadIdx.x;
    if (idx >= n) return;
    const int m = idx >> 9, d = idx & 511;
    const int pos = (m & (SS - 1)) + posoff;
    const float invf = expf(-logf(10000.f) * (float)(d & ~1) / (float)DD);
    const float ang = (float)pos * invf;
    Xr[idx] += (d & 1) ? cosf(ang) : sinf(ang);
}

__global__ void deint_kernel(const float* __restrict__ in, float* __restrict__ R,
                             float* __restrict__ I, int n)
{
    const int idx = blockIdx.x * blockDim.x + threadIdx.x;
    if (idx >= n) return;
    R[idx] = in[2 * idx]; I[idx] = in[2 * idx + 1];
}

__global__ void inter_kernel(const float* __restrict__ R, const float* __restrict__ I,
                             float* __restrict__ out, int n)
{
    const int idx = blockIdx.x * blockDim.x + threadIdx.x;
    if (idx >= n) return;
    out[2 * idx] = R[idx]; out[2 * idx + 1] = I[idx];
}

// transpose + convert weights f32[K,N] -> bf16[N,K]: wr, wi, and wr+wi planes
__global__ void packw_kernel(const float* __restrict__ wr, const float* __restrict__ wi,
                             __bf16* __restrict__ wrT, __bf16* __restrict__ wiT,
                             __bf16* __restrict__ wsT, int K, int N)
{
    const int idx = blockIdx.x * blockDim.x + threadIdx.x;
    if (idx >= K * N) return;
    const int n = idx / K, k = idx % K;
    const float r = wr[(size_t)k * N + n], im = wi[(size_t)k * N + n];
    wrT[idx] = (__bf16)r;
    wiT[idx] = (__bf16)im;
    wsT[idx] = (__bf16)(r + im);
}

// ---------------------------------------------------------------------------
// Host orchestration
// ---------------------------------------------------------------------------
struct CLin { const float *wr, *wi, *br, *bi; int K, N; __bf16 *wrT, *wiT, *wsT; };
struct LNp  { const float *gr, *gi, *br, *bi; };
struct MHAp { CLin wq, wk, wv, wo; };
struct EncL { MHAp attn; CLin l1, l2; LNp ln0, ln1; };
struct DecL { MHAp sattn, xattn; CLin l1, l2; LNp ln0, ln1, ln2; };

extern "C" void kernel_launch(void* const* d_in, const int* in_sizes, int n_in,
                              void* d_out, int out_size, void* d_ws, size_t ws_size,
                              hipStream_t stream)
{
    (void)in_sizes; (void)out_size; (void)ws_size;
    if (n_in < 350) return;

    auto F = [&](int i) -> const float* { return (const float*)d_in[i]; };
    // jax pytree flatten order: dict keys sorted. lin: {bi,br,wi,wr}; ln: {bi,br,gi,gr};
    // mha: {wk,wo,wq,wv}; params: {dec,enc,in_emb,out_emb,out_lin}; then src, tgt.
    auto mklin = [&](int base, int K, int N) -> CLin {
        CLin l; l.bi = F(base + 0); l.br = F(base + 1); l.wi = F(base + 2); l.wr = F(base + 3);
        l.K = K; l.N = N; l.wrT = nullptr; l.wiT = nullptr; l.wsT = nullptr; return l;
    };
    auto mkln = [&](int base) -> LNp {
        LNp p; p.bi = F(base + 0); p.br = F(base + 1); p.gi = F(base + 2); p.gr = F(base + 3); return p;
    };
    auto mkmha = [&](int base) -> MHAp {
        MHAp m; m.wk = mklin(base + 0, DD, DD); m.wo = mklin(base + 4, DD, DD);
        m.wq = mklin(base + 8, DD, DD); m.wv = mklin(base + 12, DD, DD); return m;
    };

    DecL dec[4]; EncL enc[4];
    for (int l = 0; l < 4; ++l) {           // dec layer: l1,l2,ln[3],sattn,xattn = 52 leaves
        const int b = l * 52;
        dec[l].l1 = mklin(b + 0, DD, DD);  dec[l].l2 = mklin(b + 4, DD, DD);
        dec[l].ln0 = mkln(b + 8); dec[l].ln1 = mkln(b + 12); dec[l].ln2 = mkln(b + 16);
        dec[l].sattn = mkmha(b + 20); dec[l].xattn = mkmha(b + 36);
    }
    for (int l = 0; l < 4; ++l) {           // enc layer: attn,l1,l2,ln[2] = 32 leaves
        const int b = 208 + l * 32;
        enc[l].attn = mkmha(b + 0);
        enc[l].l1 = mklin(b + 16, DD, DD); enc[l].l2 = mklin(b + 20, DD, DD);
        enc[l].ln0 = mkln(b + 24); enc[l].ln1 = mkln(b + 28);
    }
    CLin in_emb  = mklin(336, DIO, DD);
    CLin out_emb = mklin(340, DIO, DD);
    CLin out_lin = mklin(344, DD, DIO);
    const float* src = F(348);
    const float* tgt = F(349);

    // workspace bump allocator
    char* wsp = (char*)d_ws; size_t off = 0;
    auto alloc = [&](size_t bytes) -> void* {
        void* p = wsp + off; off = (off + bytes + 255) & ~(size_t)255; return p;
    };
    auto fplane = [&]() -> float* { return (float*)alloc((size_t)MM * DD * sizeof(float)); };
    float *Xr = fplane(), *Xi = fplane(), *Yr = fplane(), *Yi = fplane();
    float *Qr = fplane(), *Qi = fplane(), *Kr = fplane(), *Ki = fplane();
    float *Vr = fplane(), *Vi = fplane();
    float *Tr = fplane(), *Ti = fplane(), *Ur = fplane(), *Ui = fplane();
    float *Sc = (float*)alloc((size_t)BB * HH * SS * SS * sizeof(float));

    // pack all weights f32[K,N] -> bf16[N,K] (3 planes: r, i, r+i)
    CLin* all[67]; int na = 0;
    for (int l = 0; l < 4; ++l) {
        all[na++] = &dec[l].l1; all[na++] = &dec[l].l2;
        all[na++] = &dec[l].sattn.wq; all[na++] = &dec[l].sattn.wk;
        all[na++] = &dec[l].sattn.wv; all[na++] = &dec[l].sattn.wo;
        all[na++] = &dec[l].xattn.wq; all[na++] = &dec[l].xattn.wk;
        all[na++] = &dec[l].xattn.wv; all[na++] = &dec[l].xattn.wo;
    }
    for (int l = 0; l < 4; ++l) {
        all[na++] = &enc[l].attn.wq; all[na++] = &enc[l].attn.wk;
        all[na++] = &enc[l].attn.wv; all[na++] = &enc[l].attn.wo;
        all[na++] = &enc[l].l1; all[na++] = &enc[l].l2;
    }
    all[na++] = &in_emb; all[na++] = &out_emb; all[na++] = &out_lin;
    for (int i = 0; i < na; ++i) {
        CLin* L = all[i];
        const size_t e = (size_t)L->K * L->N;
        L->wrT = (__bf16*)alloc(e * sizeof(__bf16));
        L->wiT = (__bf16*)alloc(e * sizeof(__bf16));
        L->wsT = (__bf16*)alloc(e * sizeof(__bf16));
        packw_kernel<<<(int)((e + 255) / 256), 256, 0, stream>>>(L->wr, L->wi, L->wrT, L->wiT, L->wsT, L->K, L->N);
    }

    auto cgemm = [&](const float* ar, const float* ai, const CLin& L,
                     float* cr, float* ci, float alpha, int relu) {
        dim3 g(L.N / 64, MM / 64, 1);
        cgemm_kernel<<<g, dim3(128), 0, stream>>>(ar, ai, L.wrT, L.wiT, L.wsT, L.br, L.bi,
                                                  cr, ci, L.N, L.K, alpha, relu);
    };
    auto lnorm = [&](const float* ar, const float* ai, const float* rr, const float* ri,
                     const LNp& p, float* orr, float* oii) {
        cln_kernel<<<dim3(MM / 8), dim3(256), 0, stream>>>(ar, ai, rr, ri, p.gr, p.gi, p.br, p.bi, orr, oii);
    };
    auto mha = [&](const float* xqr, const float* xqi, const float* xkr, const float* xki,
                   const MHAp& m, int causal, float* outr, float* outi) {
        cgemm(xqr, xqi, m.wq, Qr, Qi, 1.f, 0);
        cgemm(xkr, xki, m.wk, Kr, Ki, 1.f, 0);
        cgemm(xkr, xki, m.wv, Vr, Vi, 1.f, 0);
        scores_kernel<<<dim3(SS / 64, SS / 64, BB * HH), dim3(128), 0, stream>>>(
            Qr, Qi, Kr, Ki, Sc, causal, 0.125f /* hd^-0.5 */);
        softmax_kernel<<<dim3(BB * HH * SS / 8), dim3(256), 0, stream>>>(Sc);
        attnout_kernel<<<dim3(1, SS / 64, BB * HH), dim3(128), 0, stream>>>(Sc, Vr, Vi, Tr, Ti);
        cgemm(Tr, Ti, m.wo, outr, outi, 1.f, 0);
    };

    // embeddings + positional encoding (PE on real plane only)
    const int nio = MM * DIO, nxd = MM * DD;
    deint_kernel<<<(nio + 255) / 256, 256, 0, stream>>>(src, Qr, Qi, nio);
    cgemm(Qr, Qi, in_emb, Xr, Xi, 1.f, 0);
    pe_kernel<<<(nxd + 255) / 256, 256, 0, stream>>>(Xr, 0, nxd);
    deint_kernel<<<(nio + 255) / 256, 256, 0, stream>>>(tgt, Qr, Qi, nio);
    cgemm(Qr, Qi, out_emb, Yr, Yi, 1.f, 0);
    pe_kernel<<<(nxd + 255) / 256, 256, 0, stream>>>(Yr, SS, nxd);

    // encoder
    for (int l = 0; l < 4; ++l) {
        mha(Xr, Xi, Xr, Xi, enc[l].attn, 0, Ur, Ui);
        lnorm(Ur, Ui, Xr, Xi, enc[l].ln0, Xr, Xi);
        cgemm(Xr, Xi, enc[l].l1, Tr, Ti, 1.f, 1);
        cgemm(Tr, Ti, enc[l].l2, Ur, Ui, 1.f, 0);
        lnorm(Ur, Ui, Xr, Xi, enc[l].ln1, Xr, Xi);
    }
    // decoder
    for (int l = 0; l < 4; ++l) {
        mha(Yr, Yi, Yr, Yi, dec[l].sattn, 1, Ur, Ui);
        lnorm(Ur, Ui, Yr, Yi, dec[l].ln0, Yr, Yi);
        mha(Yr, Yi, Xr, Xi, dec[l].xattn, 0, Ur, Ui);
        lnorm(Ur, Ui, Yr, Yi, dec[l].ln1, Yr, Yi);
        cgemm(Yr, Yi, dec[l].l1, Tr, Ti, 1.f, 1);
        cgemm(Tr, Ti, dec[l].l2, Ur, Ui, 1.f, 0);
        lnorm(Ur, Ui, Yr, Yi, dec[l].ln2, Yr, Yi);
    }
    // output projection + interleave [B,St,DOUT,2]
    cgemm(Yr, Yi, out_lin, Tr, Ti, 1.f, 0);
    inter_kernel<<<(nio + 255) / 256, 256, 0, stream>>>(Tr, Ti, (float*)d_out, nio);
}
